// BipartiteGraphConvolution_28406913696348
// MI455X (gfx1250) — matrix operand order repaired and checked by reading, not compiled, not run on hardware
//
#include <hip/hip_runtime.h>

typedef __attribute__((ext_vector_type(16))) _Float16 v16h;
typedef __attribute__((ext_vector_type(8)))  float    v8f;
typedef __attribute__((ext_vector_type(4)))  _Float16 h4;

#define EMB 128
#define LDSLD  132  // f32 staging tile row stride (floats)
#define LDSLDH 136  // f16 tile row stride (halves): 272B rows, 4-bank rotation

// ---------------------------------------------------------------------------
// Async staging: copy a 16 x 128 f32 tile (row-major, leading dim lda) into a
// padded LDS tile with CDNA5 async global->LDS copies (ASYNCcnt-tracked).
// 512 x 16B chunks, 256 threads -> 2 chunks per thread, all lanes active.
// ---------------------------------------------------------------------------
__device__ inline void stage_tile_async(const float* __restrict__ g, int row0,
                                        int lda, float* lds, int tid) {
  for (int c = tid; c < 512; c += 256) {
    int r = c >> 5;        // row in tile
    int w = c & 31;        // 16B chunk within row
    unsigned lds_addr =
        (unsigned)(uintptr_t)(lds + r * LDSLD + w * 4);
    unsigned long long gaddr =
        (unsigned long long)(uintptr_t)(g + (size_t)(row0 + r) * lda + w * 4);
    asm volatile("global_load_async_to_lds_b128 %0, %1, off"
                 :: "v"(lds_addr), "v"(gaddr) : "memory");
  }
}

__device__ inline void wait_async_all() {
  asm volatile("s_wait_asynccnt 0x0" ::: "memory");
}

// Cooperative one-shot f32->f16 tile conversion in LDS (512 float4 chunks).
__device__ inline void cvt_tile_lds(const float* stage, _Float16* dst, int tid) {
  for (int i = tid; i < 512; i += 256) {
    int r  = i >> 5;
    int c4 = (i & 31) * 4;
    float4 v = *(const float4*)&stage[r * LDSLD + c4];
    h4 o;
    o[0] = (_Float16)v.x; o[1] = (_Float16)v.y;
    o[2] = (_Float16)v.z; o[3] = (_Float16)v.w;
    *(h4*)&dst[r * LDSLDH + c4] = o;
  }
}

// ---------------------------------------------------------------------------
// WMMA fragment loaders (wave32 layouts, CDNA5 ISA 7.12.2)
// ---------------------------------------------------------------------------

// A-matrix 16x32 f16 fragment from an f16 LDS tile with row stride LDSLDH.
// lane<16: row=lane,     elems 0..7 -> K=k0+0..7,  elems 8..15 -> K=k0+16..23
// lane>=16: row=lane-16, elems 0..7 -> K=k0+8..15, elems 8..15 -> K=k0+24..31
__device__ inline v16h load_a_frag_ldsh(const _Float16* A, int k0, int lane) {
  int r  = lane & 15;
  int kb = k0 + ((lane & 16) ? 8 : 0);
  const _Float16* p = A + r * LDSLDH + kb;
  v16h a;
#pragma unroll
  for (int j = 0; j < 8; ++j) {
    a[j]     = p[j];
    a[j + 8] = p[j + 16];
  }
  return a;
}

// B-matrix 32x16 fragment representing W^T from a pre-converted f16 weight
// matrix (row-major, leading dim ldw): B[k][n] = W[n][k].
// lane<16: col n=n0+lane, elems 0..7 -> K=k0+0..7, 8..15 -> K=k0+16..23; lane>=16: K=k0+8..15, k0+24..31.
__device__ inline v16h load_b_frag_h(const _Float16* __restrict__ Wh, int ldw,
                                     int n0, int k0, int lane) {
  int n  = n0 + (lane & 15);
  int kb = k0 + ((lane & 16) ? 8 : 0);
  const _Float16* p = Wh + (size_t)n * ldw + kb;
  v16h b;
#pragma unroll
  for (int j = 0; j < 8; ++j) { b[j] = p[j]; b[j + 8] = p[j + 16]; }
  return b;
}

// C/D f32 16x16: VGPR j -> (M = j + (lane>=16 ? 8:0), N = lane&15)

// ---------------------------------------------------------------------------
// Prep: convert f32 weights to f16 once
// ---------------------------------------------------------------------------
__global__ void cvt_kernel(const float* __restrict__ src,
                           _Float16* __restrict__ dst, int n) {
  int i = blockIdx.x * blockDim.x + threadIdx.x;
  if (i < n) dst[i] = (_Float16)src[i];
}

__global__ void zero_kernel(float* __restrict__ p, size_t n) {
  size_t i = (size_t)blockIdx.x * blockDim.x + threadIdx.x;
  if (i < n) p[i] = 0.0f;
}

// ---------------------------------------------------------------------------
// Kernel 1/2: Yh[nrows,128] = f16( X @ W^T (+ bias) ).
// A tile: async stage -> cooperative f16 convert -> direct f16 fragments.
// Wave w computes cols 16w..16w+15.
// ---------------------------------------------------------------------------
__global__ __launch_bounds__(256)
void xform_kernel(const float* __restrict__ X, const _Float16* __restrict__ Wh,
                  const float* __restrict__ bias, _Float16* __restrict__ Y,
                  int nrows) {
  __shared__ alignas(16) float    stage[16 * LDSLD];
  __shared__ alignas(16) _Float16 ah[16 * LDSLDH];
  (void)nrows;
  int tid  = threadIdx.x;
  int lane = tid & 31;
  int wv   = tid >> 5;
  int row0 = blockIdx.x * 16;

  stage_tile_async(X, row0, EMB, stage, tid);
  wait_async_all();
  __syncthreads();
  cvt_tile_lds(stage, ah, tid);
  __syncthreads();

  int n0 = wv * 16;
  v8f acc = {};
#pragma unroll
  for (int kk = 0; kk < EMB; kk += 32) {
    v16h a = load_a_frag_ldsh(ah, kk, lane);
    v16h b = load_b_frag_h(Wh, EMB, n0, kk, lane);
    acc = __builtin_amdgcn_wmma_f32_16x16x32_f16(false, a, false, b,
                                                 (short)0, acc, false, false);
  }
  int n  = n0 + (lane & 15);
  int mb = (lane & 16) ? 8 : 0;
  float bn = bias ? bias[n] : 0.0f;
#pragma unroll
  for (int j = 0; j < 8; ++j)
    Y[(size_t)(row0 + mb + j) * EMB + n] = (_Float16)(acc[j] + bn);
}

// ---------------------------------------------------------------------------
// Edge phase: one wave per edge; lane handles 4 channels (f16 gathers,
// f32 math, f32 atomic scatter).
// S[dst] += relu((Lt[src] + ef*W_edge + Rt[dst]) * s_pre);  cnt[dst] += 1
// ---------------------------------------------------------------------------
__global__ __launch_bounds__(256)
void edge_kernel(const _Float16* __restrict__ Lth, const _Float16* __restrict__ Rth,
                 const int* __restrict__ eidx, const float* __restrict__ ef,
                 const float* __restrict__ Wedge, const float* __restrict__ s_pre,
                 float* __restrict__ S, float* __restrict__ cnt, int E) {
  int lane = threadIdx.x & 31;
  int e = blockIdx.x * (blockDim.x >> 5) + (threadIdx.x >> 5);
  if (e >= E) return;

  int src = eidx[e];
  int dst = eidx[E + e];
  float sp = s_pre[0];
  float ev = ef[e];

  const h4*     l4 = (const h4*)(Lth + (size_t)src * EMB);
  const h4*     r4 = (const h4*)(Rth + (size_t)dst * EMB);
  const float4* w4 = (const float4*)Wedge;
  h4     lv = l4[lane];
  h4     rv = r4[lane];
  float4 wv = w4[lane];

  float4 v;
  v.x = fmaxf(((float)lv[0] + ev * wv.x + (float)rv[0]) * sp, 0.0f);
  v.y = fmaxf(((float)lv[1] + ev * wv.y + (float)rv[1]) * sp, 0.0f);
  v.z = fmaxf(((float)lv[2] + ev * wv.z + (float)rv[2]) * sp, 0.0f);
  v.w = fmaxf(((float)lv[3] + ev * wv.w + (float)rv[3]) * sp, 0.0f);

  float* Sd = S + (size_t)dst * EMB + lane * 4;
  atomicAdd(Sd + 0, v.x);
  atomicAdd(Sd + 1, v.y);
  atomicAdd(Sd + 2, v.z);
  atomicAdd(Sd + 3, v.w);
  if (lane == 0) atomicAdd(&cnt[dst], 1.0f);
}

// ---------------------------------------------------------------------------
// Fused tail per 16-row tile of right nodes (S + right tiles staged async,
// converted to f16 once; intermediates kept as f16 LDS tiles):
//   conv = (S @ Wf^T + cnt (x) bf) * s_post          -> cA (f16 LDS)
//   h    = relu(conv @ W1a^T + right @ W1b^T + b1)   -> hA (f16 LDS)
//   out  = h @ W2^T + b2                             -> global f32
// W1h is f16 W_out1 [128 x 256] row-major: W1a = cols 0..127, W1b = 128..255.
// ---------------------------------------------------------------------------
__global__ __launch_bounds__(256)
void out_kernel(const float* __restrict__ S, const float* __restrict__ cnt,
                const float* __restrict__ right,
                const _Float16* __restrict__ Wfh, const float* __restrict__ bf,
                const float* __restrict__ s_post,
                const _Float16* __restrict__ W1h, const float* __restrict__ b1,
                const _Float16* __restrict__ W2h, const float* __restrict__ b2,
                float* __restrict__ out) {
  __shared__ alignas(16) float    stg0[16 * LDSLD];
  __shared__ alignas(16) float    stg1[16 * LDSLD];
  __shared__ alignas(16) _Float16 sA[16 * LDSLDH];  // f16 S tile
  __shared__ alignas(16) _Float16 rA[16 * LDSLDH];  // f16 right tile
  __shared__ alignas(16) _Float16 cA[16 * LDSLDH];  // f16 conv tile
  __shared__ alignas(16) _Float16 hA[16 * LDSLDH];  // f16 h tile

  int tid  = threadIdx.x;
  int lane = tid & 31;
  int wv   = tid >> 5;
  int r0   = blockIdx.x * 16;
  int n0   = wv * 16;
  int n    = n0 + (lane & 15);
  int mb   = (lane & 16) ? 8 : 0;
  float spost = s_post[0];

  stage_tile_async(S, r0, EMB, stg0, tid);
  stage_tile_async(right, r0, EMB, stg1, tid);
  wait_async_all();
  __syncthreads();
  cvt_tile_lds(stg0, sA, tid);
  cvt_tile_lds(stg1, rA, tid);
  __syncthreads();

  // ---- conv tile (reads sA, writes cA as f16) ----
  v8f acc0 = {};
#pragma unroll
  for (int kk = 0; kk < EMB; kk += 32) {
    v16h a = load_a_frag_ldsh(sA, kk, lane);
    v16h b = load_b_frag_h(Wfh, EMB, n0, kk, lane);
    acc0 = __builtin_amdgcn_wmma_f32_16x16x32_f16(false, a, false, b,
                                                  (short)0, acc0, false, false);
  }
  float bfn = bf[n];
#pragma unroll
  for (int j = 0; j < 8; ++j) {
    int m = mb + j;
    cA[m * LDSLDH + n] = (_Float16)((acc0[j] + cnt[r0 + m] * bfn) * spost);
  }
  __syncthreads();

  // ---- h tile (reads cA + rA, writes hA as f16) ----
  v8f acc1 = {};
#pragma unroll
  for (int kk = 0; kk < EMB; kk += 32) {
    v16h a = load_a_frag_ldsh(cA, kk, lane);
    v16h b = load_b_frag_h(W1h, 2 * EMB, n0, kk, lane);        // W1a
    acc1 = __builtin_amdgcn_wmma_f32_16x16x32_f16(false, a, false, b,
                                                  (short)0, acc1, false, false);
  }
#pragma unroll
  for (int kk = 0; kk < EMB; kk += 32) {
    v16h a = load_a_frag_ldsh(rA, kk, lane);
    v16h b = load_b_frag_h(W1h + EMB, 2 * EMB, n0, kk, lane);  // W1b
    acc1 = __builtin_amdgcn_wmma_f32_16x16x32_f16(false, a, false, b,
                                                  (short)0, acc1, false, false);
  }
  float b1n = b1[n];
#pragma unroll
  for (int j = 0; j < 8; ++j)
    hA[(mb + j) * LDSLDH + n] = (_Float16)fmaxf(acc1[j] + b1n, 0.0f);
  __syncthreads();

  // ---- out tile (reads hA) ----
  v8f acc2 = {};
#pragma unroll
  for (int kk = 0; kk < EMB; kk += 32) {
    v16h a = load_a_frag_ldsh(hA, kk, lane);
    v16h b = load_b_frag_h(W2h, EMB, n0, kk, lane);
    acc2 = __builtin_amdgcn_wmma_f32_16x16x32_f16(false, a, false, b,
                                                  (short)0, acc2, false, false);
  }
  float b2n = b2[n];
#pragma unroll
  for (int j = 0; j < 8; ++j)
    out[(size_t)(r0 + mb + j) * EMB + n] = acc2[j] + b2n;
}

// ---------------------------------------------------------------------------
// Launch
// ---------------------------------------------------------------------------
extern "C" void kernel_launch(void* const* d_in, const int* in_sizes, int n_in,
                              void* d_out, int out_size, void* d_ws, size_t ws_size,
                              hipStream_t stream) {
  const float* left    = (const float*)d_in[0];
  const int*   eidx    = (const int*)d_in[1];
  const float* ef      = (const float*)d_in[2];
  const float* right   = (const float*)d_in[3];
  (void)d_in[4]; // scatter_out_size (derived from in_sizes[3])
  const float* W_left  = (const float*)d_in[5];
  const float* b_left  = (const float*)d_in[6];
  const float* W_edge  = (const float*)d_in[7];
  const float* W_right = (const float*)d_in[8];
  const float* s_pre   = (const float*)d_in[9];
  const float* W_final = (const float*)d_in[10];
  const float* b_final = (const float*)d_in[11];
  const float* s_post  = (const float*)d_in[12];
  const float* W_out1  = (const float*)d_in[13];
  const float* b_out1  = (const float*)d_in[14];
  const float* W_out2  = (const float*)d_in[15];
  const float* b_out2  = (const float*)d_in[16];

  const int NL = in_sizes[0] / EMB;   // 50000
  const int NR = in_sizes[3] / EMB;   // 100000
  const int E  = in_sizes[2];         // 500000

  // Workspace layout (all regions 256B-aligned)
  float* Sb = (float*)d_ws;                       // NR*EMB f32 (scatter acc)
  float* cn = Sb + (size_t)NR * EMB;              // NR f32 (edge counts)
  size_t cnpad = ((size_t)NR + 127) & ~(size_t)127;
  _Float16* Lth = (_Float16*)(cn + cnpad);        // NL*EMB f16
  _Float16* Rth = Lth + (size_t)NL * EMB;         // NR*EMB f16
  _Float16* Wlh = Rth + (size_t)NR * EMB;         // 128x128
  _Float16* Wrh = Wlh + EMB * EMB;                // 128x128
  _Float16* Wfh = Wrh + EMB * EMB;                // 128x128
  _Float16* W1h = Wfh + EMB * EMB;                // 128x256
  _Float16* W2h = W1h + 2 * EMB * EMB;            // 128x128

  // 0) weight conversion to f16 (once)
  cvt_kernel<<<(EMB * EMB + 255) / 256, 256, 0, stream>>>(W_left, Wlh, EMB * EMB);
  cvt_kernel<<<(EMB * EMB + 255) / 256, 256, 0, stream>>>(W_right, Wrh, EMB * EMB);
  cvt_kernel<<<(EMB * EMB + 255) / 256, 256, 0, stream>>>(W_final, Wfh, EMB * EMB);
  cvt_kernel<<<(2 * EMB * EMB + 255) / 256, 256, 0, stream>>>(W_out1, W1h, 2 * EMB * EMB);
  cvt_kernel<<<(EMB * EMB + 255) / 256, 256, 0, stream>>>(W_out2, W2h, EMB * EMB);

  // 1) Lt = f16(left @ W_left^T + b_left)
  xform_kernel<<<(NL + 15) / 16, 256, 0, stream>>>(left, Wlh, b_left, Lth, NL);
  // 2) Rt = f16(right @ W_right^T)
  xform_kernel<<<(NR + 15) / 16, 256, 0, stream>>>(right, Wrh, nullptr, Rth, NR);
  // 3) zero scatter accumulators + counts
  size_t zn = (size_t)NR * EMB + cnpad;
  zero_kernel<<<(unsigned)((zn + 255) / 256), 256, 0, stream>>>(Sb, zn);
  // 4) edge scatter (wave per edge)
  int epb = 256 / 32;
  edge_kernel<<<(E + epb - 1) / epb, 256, 0, stream>>>(Lth, Rth, eidx, ef, W_edge,
                                                       s_pre, Sb, cn, E);
  // 5) fused conv + output MLP
  out_kernel<<<(NR + 15) / 16, 256, 0, stream>>>(Sb, cn, right, Wfh, b_final,
                                                 s_post, W1h, b_out1,
                                                 W2h, b_out2, (float*)d_out);
}
